// SSM_2920577761702
// MI455X (gfx1250) — matrix operand order repaired and checked by reading, not compiled, not run on hardware
//
#include <hip/hip_runtime.h>

// Problem constants
#define Bsz 4
#define Lsz 4096
#define Dsz 1024
#define Nsz 1024
#define Msz (Bsz * Lsz)   // 16384 rows
#define NC  64            // scan chunks
#define CL  64            // steps per chunk (NC*CL == Lsz)

// WMMA fragment types (gfx1250, wave32)
typedef __attribute__((ext_vector_type(16))) __bf16 v16bf;
typedef __attribute__((ext_vector_type(8)))  float  v8f;

union FragBF { unsigned int u[8]; v16bf v; };

// TDM descriptor group types (6-arg builtin: g0, g1, g2, g3, g4, cpol)
typedef __attribute__((ext_vector_type(4))) unsigned int tdm_g0_t;
typedef __attribute__((ext_vector_type(8))) int          tdm_g1_t;
typedef __attribute__((ext_vector_type(4))) int          tdm_g2_t;

// LDS tile strides (elements); 16B-aligned rows
#define LDA 40
#define LDB 72
#define KSTEPS (Dsz / 32)

// ---------- helpers ----------
__device__ __forceinline__ unsigned short f2bf(float f) {
  unsigned int u = __float_as_uint(f);
  unsigned int r = u + 0x7FFFu + ((u >> 16) & 1u);
  return (unsigned short)(r >> 16);
}
__device__ __forceinline__ float bf2f(unsigned short h) {
  return __uint_as_float(((unsigned int)h) << 16);
}
__device__ __forceinline__ float sigm(float x) {
  return 1.0f / (1.0f + __expf(-x));
}

// CDNA5 async copy global -> LDS (16B per lane), tracked by ASYNCcnt
__device__ __forceinline__ void async_cp16(unsigned ldsoff, const void* g) {
  asm volatile("global_load_async_to_lds_b128 %0, %1, off"
               :: "v"(ldsoff), "v"((unsigned long long)(size_t)g)
               : "memory");
}
__device__ __forceinline__ void wait_async2() {
  asm volatile("s_wait_asynccnt 0x2" ::: "memory");
}
__device__ __forceinline__ void wait_async0() {
  asm volatile("s_wait_asynccnt 0x0" ::: "memory");
}

// CDNA5 Tensor Data Mover: load 32x64 bf16 tile (row stride 1024 elems)
// into LDS with padded row stride LDB=72 elems (pad 4 DWORDs per 32 DWORDs).
__device__ __forceinline__ void tdm_load_B(unsigned lds_addr, const void* g) {
  unsigned long long ga = (unsigned long long)(size_t)g;
  tdm_g0_t g0 = { 1u,                                    // count=1 (valid D#)
                  lds_addr,                              // LDS byte address
                  (unsigned int)ga,                      // global addr [31:0]
                  (unsigned int)((ga >> 32) & 0x01FFFFFFu) | 0x80000000u }; // [56:32] | type=2
  // group1: wg_mask=0, data_size=1(2B), pad_enable=1, pad_interval=4(32 DW),
  //         pad_amount=3(4 DW); tensor_dim0=1024, tensor_dim1=1024,
  //         tile_dim0=64, tile_dim1=32, tile_dim2=0, dim0_stride=1024
  tdm_g1_t g1 = { (int)0x07110000, (int)0x04000000, (int)0x04000000,
                  (int)0x00400000, 32, 1024, 0, 0 };
  tdm_g2_t z4 = { 0, 0, 0, 0 };
  tdm_g1_t z8 = { 0, 0, 0, 0, 0, 0, 0, 0 };
  __builtin_amdgcn_tensor_load_to_lds(g0, g1, z4, z4, z8, 0);
}

// ---------- 1) weight fp32 -> bf16 conversion ----------
__global__ void __launch_bounds__(256) cvt_w_kernel(
    const float* __restrict__ wA, const float* __restrict__ wx,
    const float* __restrict__ wsg, const float* __restrict__ wout,
    const float* __restrict__ wo,
    unsigned short* __restrict__ wcat, unsigned short* __restrict__ wob) {
  unsigned int id = blockIdx.x * 256u + threadIdx.x;      // 5 * 2^20 total
  if (id < (4u << 20)) {
    unsigned int mat = id >> 20;
    unsigned int off = id & ((1u << 20) - 1u);
    const float* s = (mat == 0) ? wA : (mat == 1) ? wx : (mat == 2) ? wsg : wout;
    wcat[id] = f2bf(s[off]);
  } else {
    unsigned int off = id - (4u << 20);
    wob[off] = f2bf(wo[off]);
  }
}

// ---------- 2) RMSNorm -> bf16 ----------
__global__ void __launch_bounds__(256) rmsnorm_kernel(
    const float* __restrict__ x, const float* __restrict__ w,
    unsigned short* __restrict__ xn) {
  __shared__ float red[8];
  const int row = blockIdx.x;
  const int c = threadIdx.x * 4;
  const float* xr = x + (size_t)row * Dsz;
  float4 d = *(const float4*)(xr + c);
  float ss = d.x * d.x + d.y * d.y + d.z * d.z + d.w * d.w;
  for (int off = 16; off > 0; off >>= 1) ss += __shfl_down(ss, off, 32);
  if ((threadIdx.x & 31) == 0) red[threadIdx.x >> 5] = ss;
  __syncthreads();
  if (threadIdx.x == 0) {
    float s = 0.f;
    for (int i = 0; i < 8; i++) s += red[i];
    red[0] = s;
  }
  __syncthreads();
  const float inv = rsqrtf(red[0] * (1.0f / (float)Dsz) + 1e-5f);
  float4 wv = *(const float4*)(w + c);
  unsigned int p0 = (unsigned int)f2bf(d.x * wv.x * inv) |
                    ((unsigned int)f2bf(d.y * wv.y * inv) << 16);
  unsigned int p1 = (unsigned int)f2bf(d.z * wv.z * inv) |
                    ((unsigned int)f2bf(d.w * wv.w * inv) << 16);
  uint2 pk; pk.x = p0; pk.y = p1;
  *(uint2*)(xn + (size_t)row * Dsz + c) = pk;
}

// ---------- shared WMMA 128x64 tile body ----------
// A tile: async-to-LDS per thread; B tile: one TDM op per block (wave 0).
__device__ __forceinline__ void gemm_tile_128x64(
    const unsigned short* __restrict__ Ag,   // A rows, base at mblk, ld = 1024
    const unsigned short* __restrict__ Bg,   // B with col base applied, ld = 1024
    unsigned short (*As)[128 * LDA],         // As[2][...]
    unsigned short (*Bs)[32 * LDB],          // Bs[2][...]
    v8f acc[2][2]) {
  const int tid = threadIdx.x;
  const int lane = tid & 31;
  const int wave = tid >> 5;
  const int wm = wave & 3;   // 4 waves along M (32 rows each)
  const int wn = wave >> 2;  // 2 waves along N (32 cols each)
  const int mlane = lane & 15;
  const int kh = lane >> 4;
  const bool issuer = (tid < 32);   // wave-uniform branch; TDM issued once/block

  // per-thread A staging addresses (128x32 tile, 16 elems/thread)
  const int arow = tid >> 1, akc = (tid & 1) * 16;
  const unsigned short* aSrc = Ag + (size_t)arow * Dsz + akc;

  unsigned aoff0[2], aoff1[2], bsbase[2];
  for (int b = 0; b < 2; b++) {
    aoff0[b] = (unsigned)(size_t)&As[b][arow * LDA + akc];
    aoff1[b] = aoff0[b] + 16;   // +8 elements
    bsbase[b] = (unsigned)(size_t)&Bs[b][0];
  }

  // prologue: stage 0
  async_cp16(aoff0[0], aSrc);
  async_cp16(aoff1[0], aSrc + 8);
  if (issuer) tdm_load_B(bsbase[0], Bg);

  for (int kk = 0; kk < KSTEPS; ++kk) {
    const int buf = kk & 1;
    if (kk + 1 < KSTEPS) {
      const int k0n = (kk + 1) * 32;
      async_cp16(aoff0[buf ^ 1], aSrc + k0n);
      async_cp16(aoff1[buf ^ 1], aSrc + k0n + 8);
      if (issuer) tdm_load_B(bsbase[buf ^ 1], Bg + (size_t)k0n * Nsz);
      wait_async2();                              // stage kk's A copies done
      if (issuer) __builtin_amdgcn_s_wait_tensorcnt(1);  // stage kk's B done
    } else {
      wait_async0();
      if (issuer) __builtin_amdgcn_s_wait_tensorcnt(0);
    }
    __syncthreads();   // all waves' copies for stage kk visible

    FragBF afr[2], bfr[2];
    for (int tm = 0; tm < 2; tm++) {
      int Mr = wm * 32 + tm * 16 + mlane;
      for (int r = 0; r < 8; r++) {
        int K0 = (r >> 2) * 16 + kh * 8 + (r & 3) * 2;
        afr[tm].u[r] = *(const unsigned int*)&As[buf][Mr * LDA + K0];
      }
    }
    for (int tn = 0; tn < 2; tn++) {
      int Ncb = wn * 32 + tn * 16;
      for (int r = 0; r < 8; r++) {
        int K0 = (r >> 2) * 16 + kh * 8 + (r & 3) * 2;
        bfr[tn].u[r] = (unsigned int)Bs[buf][K0 * LDB + Ncb + mlane] |
                       ((unsigned int)Bs[buf][(K0 + 1) * LDB + Ncb + mlane] << 16);
      }
    }
    for (int tm = 0; tm < 2; tm++)
      for (int tn = 0; tn < 2; tn++)
        acc[tm][tn] = __builtin_amdgcn_wmma_f32_16x16x32_bf16(
            false, afr[tm].v, false, bfr[tn].v, (short)0, acc[tm][tn], false, false);
    __syncthreads();   // done reading buf before it is overwritten
  }
}

// ---------- 3) fused 4-way projection GEMM (WMMA bf16) ----------
__global__ void __launch_bounds__(256) gemm_proj_kernel(
    const unsigned short* __restrict__ xn,    // [Msz][Dsz] bf16
    const unsigned short* __restrict__ wcat,  // 4 x [Dsz][Nsz] bf16
    const float* __restrict__ bA,
    float* __restrict__ Abuf,                 // [Msz][Nsz] f32
    unsigned short* __restrict__ Xel,         // bf16
    unsigned short* __restrict__ Gate,        // bf16 (sigmoid(xn@w_sidegate))
    unsigned short* __restrict__ Sg) {        // bf16 (sigmoid(xn@w_output))
  __shared__ __align__(16) unsigned short As[2][128 * LDA];
  __shared__ __align__(16) unsigned short Bs[2][32 * LDB];

  const int lane = threadIdx.x & 31;
  const int wave = threadIdx.x >> 5;
  const int wm = wave & 3;
  const int wn = wave >> 2;
  const int mblk = blockIdx.x * 128;
  const int nbcat = blockIdx.y;          // 0..63
  const int mat = nbcat >> 4;            // which weight matrix
  const int n0 = (nbcat & 15) * 64;      // column base within that matrix

  v8f acc[2][2];
  for (int i = 0; i < 2; i++)
    for (int j = 0; j < 2; j++)
      for (int r = 0; r < 8; r++) acc[i][j][r] = 0.0f;

  gemm_tile_128x64(xn + (size_t)mblk * Dsz,
                   wcat + ((size_t)mat << 20) + n0, As, Bs, acc);

  const int mlane = lane & 15;
  const int half = lane >> 4;
  for (int tm = 0; tm < 2; tm++) {
    for (int tn = 0; tn < 2; tn++) {
      int col = n0 + wn * 32 + tn * 16 + mlane;
      for (int r = 0; r < 8; r++) {
        int row = mblk + wm * 32 + tm * 16 + half * 8 + r;
        float v = acc[tm][tn][r];
        size_t idx = (size_t)row * Nsz + col;
        if (mat == 0) {
          Abuf[idx] = sigm(v + bA[col]);        // decay gate A (fp32 for scan)
        } else if (mat == 1) {
          Xel[idx] = f2bf(v);                   // raw input
        } else if (mat == 2) {
          Gate[idx] = f2bf(sigm(v));            // output gate (w_sidegate)
        } else {
          Sg[idx] = f2bf(sigm(v));              // input gate (w_output)
        }
      }
    }
  }
}

// ---------- 4) scan pass 1: per-chunk (a_prod, h_end) ----------
__global__ void __launch_bounds__(256) scan1_kernel(
    const float* __restrict__ Abuf, const unsigned short* __restrict__ Xel,
    const unsigned short* __restrict__ Sg,
    float* __restrict__ cA, float* __restrict__ cH) {
  const int n = blockIdx.x * 256 + threadIdx.x;
  const int c = blockIdx.y, b = blockIdx.z;
  size_t base = ((size_t)b * Lsz + (size_t)c * CL) * Nsz + n;
  float ap = 1.0f, h = 0.0f;
  for (int i = 0; i < CL; i++) {
    size_t idx = base + (size_t)i * Nsz;
    float a = Abuf[idx];
    float u = bf2f(Xel[idx]) * bf2f(Sg[idx]);
    h = fmaf(a, h, u);
    ap *= a;
  }
  size_t cidx = ((size_t)b * NC + c) * Nsz + n;
  cA[cidx] = ap;
  cH[cidx] = h;
}

// ---------- 5) scan pass 2: chunk-carry scan ----------
__global__ void __launch_bounds__(256) scan2_kernel(
    const float* __restrict__ cA, const float* __restrict__ cH,
    float* __restrict__ cin) {
  const int n = blockIdx.x * 256 + threadIdx.x;
  const int b = blockIdx.y;
  float h = 0.0f;
  for (int c = 0; c < NC; c++) {
    size_t idx = ((size_t)b * NC + c) * Nsz + n;
    cin[idx] = h;
    h = fmaf(cA[idx], h, cH[idx]);
  }
}

// ---------- 6) scan pass 3: replay with carry, gate, emit y (bf16) ----------
__global__ void __launch_bounds__(256) scan3_kernel(
    const float* __restrict__ Abuf, const unsigned short* __restrict__ Xel,
    const unsigned short* __restrict__ Sg, const unsigned short* __restrict__ Gate,
    const float* __restrict__ cin, unsigned short* __restrict__ Y) {
  const int n = blockIdx.x * 256 + threadIdx.x;
  const int c = blockIdx.y, b = blockIdx.z;
  float h = cin[((size_t)b * NC + c) * Nsz + n];
  size_t base = ((size_t)b * Lsz + (size_t)c * CL) * Nsz + n;
  for (int i = 0; i < CL; i++) {
    size_t idx = base + (size_t)i * Nsz;
    float a = Abuf[idx];
    float u = bf2f(Xel[idx]) * bf2f(Sg[idx]);
    h = fmaf(a, h, u);
    Y[idx] = f2bf(bf2f(Gate[idx]) * h);
  }
}

// ---------- 7) output GEMM (WMMA bf16) + residual ----------
__global__ void __launch_bounds__(256) gemm_out_kernel(
    const unsigned short* __restrict__ Y,   // [Msz][Nsz] bf16
    const unsigned short* __restrict__ Wo,  // [Nsz][Dsz] bf16
    const float* __restrict__ xres,         // residual
    float* __restrict__ out) {
  __shared__ __align__(16) unsigned short As[2][128 * LDA];
  __shared__ __align__(16) unsigned short Bs[2][32 * LDB];

  const int lane = threadIdx.x & 31;
  const int wave = threadIdx.x >> 5;
  const int wm = wave & 3;
  const int wn = wave >> 2;
  const int mblk = blockIdx.x * 128;
  const int n0 = blockIdx.y * 64;

  v8f acc[2][2];
  for (int i = 0; i < 2; i++)
    for (int j = 0; j < 2; j++)
      for (int r = 0; r < 8; r++) acc[i][j][r] = 0.0f;

  gemm_tile_128x64(Y + (size_t)mblk * Nsz, Wo + n0, As, Bs, acc);

  const int mlane = lane & 15;
  const int half = lane >> 4;
  for (int tm = 0; tm < 2; tm++) {
    for (int tn = 0; tn < 2; tn++) {
      int col = n0 + wn * 32 + tn * 16 + mlane;
      for (int r = 0; r < 8; r++) {
        int row = mblk + wm * 32 + tm * 16 + half * 8 + r;
        size_t idx = (size_t)row * Dsz + col;
        out[idx] = acc[tm][tn][r] + xres[idx];
      }
    }
  }
}

// ---------- workspace layout (bytes) ----------
#define OFF_WCAT ((size_t)0)              //  8 MiB: 4 x 1M bf16
#define OFF_WO   ((size_t)8  << 20)       //  2 MiB
#define OFF_XN   ((size_t)10 << 20)       // 32 MiB
#define OFF_A    ((size_t)42 << 20)       // 64 MiB f32
#define OFF_XEL  ((size_t)106 << 20)      // 32 MiB bf16
#define OFF_SG   ((size_t)138 << 20)      // 32 MiB bf16
#define OFF_GATE ((size_t)170 << 20)      // 32 MiB bf16
#define OFF_CA   ((size_t)202 << 20)      //  1 MiB
#define OFF_CH   ((size_t)203 << 20)      //  1 MiB
#define OFF_CIN  ((size_t)204 << 20)      //  1 MiB
#define OFF_Y    ((size_t)205 << 20)      // 32 MiB bf16

extern "C" void kernel_launch(void* const* d_in, const int* in_sizes, int n_in,
                              void* d_out, int out_size, void* d_ws, size_t ws_size,
                              hipStream_t stream) {
  const float* x    = (const float*)d_in[0];
  const float* rmsw = (const float*)d_in[1];
  const float* wA   = (const float*)d_in[2];
  const float* bA   = (const float*)d_in[3];
  const float* wx   = (const float*)d_in[4];
  const float* wsg  = (const float*)d_in[5];
  const float* wout = (const float*)d_in[6];
  const float* wo   = (const float*)d_in[7];

  char* ws = (char*)d_ws;
  unsigned short* wcat = (unsigned short*)(ws + OFF_WCAT);
  unsigned short* wob  = (unsigned short*)(ws + OFF_WO);
  unsigned short* xn   = (unsigned short*)(ws + OFF_XN);
  float*          Abuf = (float*)(ws + OFF_A);
  unsigned short* Xel  = (unsigned short*)(ws + OFF_XEL);
  unsigned short* Sg   = (unsigned short*)(ws + OFF_SG);
  unsigned short* Gate = (unsigned short*)(ws + OFF_GATE);
  float*          cA   = (float*)(ws + OFF_CA);
  float*          cH   = (float*)(ws + OFF_CH);
  float*          cin  = (float*)(ws + OFF_CIN);
  unsigned short* Y    = (unsigned short*)(ws + OFF_Y);
  float* out = (float*)d_out;

  // 1) weights -> bf16 (5 * 2^20 elements)
  cvt_w_kernel<<<(5u << 20) / 256, 256, 0, stream>>>(wA, wx, wsg, wout, wo, wcat, wob);
  // 2) RMSNorm
  rmsnorm_kernel<<<Msz, 256, 0, stream>>>(x, rmsw, xn);
  // 3) fused projections (M/128 x 4N/64 blocks)
  gemm_proj_kernel<<<dim3(Msz / 128, 64), 256, 0, stream>>>(xn, wcat, bA, Abuf, Xel, Gate, Sg);
  // 4-6) chunked scan
  scan1_kernel<<<dim3(Nsz / 256, NC, Bsz), 256, 0, stream>>>(Abuf, Xel, Sg, cA, cH);
  scan2_kernel<<<dim3(Nsz / 256, Bsz), 256, 0, stream>>>(cA, cH, cin);
  scan3_kernel<<<dim3(Nsz / 256, NC, Bsz), 256, 0, stream>>>(Abuf, Xel, Sg, Gate, cin, Y);
  // 7) output GEMM + residual
  gemm_out_kernel<<<dim3(Msz / 128, Dsz / 64), 256, 0, stream>>>(Y, wob, x, out);
}
